// ReprogrammingLayer_11965778886798
// MI455X (gfx1250) — matrix operand order repaired and checked by reading, not compiled, not run on hardware
//
#include <hip/hip_runtime.h>
#include <hip/hip_bf16.h>
#include <math.h>

typedef __bf16 bf16;
typedef __attribute__((ext_vector_type(16))) __bf16 v16bf;
typedef __attribute__((ext_vector_type(8)))  float  v8f;
typedef __attribute__((vector_size(16)))     int    vi4;   // 16-byte chunk type

#define WMMA_BF16(a, b, c) \
  __builtin_amdgcn_wmma_f32_16x16x32_bf16(false, (a), false, (b), (short)0, (c), false, false)

// ---- async global -> LDS path (CDNA5 GLOBAL_LOAD_ASYNC_TO_LDS_B128) ----
#if defined(__has_builtin)
#  if __has_builtin(__builtin_amdgcn_global_load_async_to_lds_b128)
#    define ASYNC_LDS 1
#  endif
#endif
#ifndef ASYNC_LDS
#  define ASYNC_LDS 0
#endif

#if ASYNC_LDS
#  if __has_builtin(__builtin_amdgcn_s_wait_asynccnt)
#    define ASYNC_WAIT() __builtin_amdgcn_s_wait_asynccnt(0)
#  else
#    define ASYNC_WAIT() asm volatile("s_wait_asynccnt 0" ::: "memory")
#  endif
#else
#  define ASYNC_WAIT() ((void)0)
#endif

__device__ __forceinline__ void copy16_to_lds(void* lds, const void* g) {
#if ASYNC_LDS
  __builtin_amdgcn_global_load_async_to_lds_b128(
      (__attribute__((address_space(1))) vi4*)g,
      (__attribute__((address_space(3))) vi4*)lds, 0, 0);
#else
  *(uint4*)lds = *(const uint4*)g;
#endif
}

__device__ __forceinline__ uint2 pack_bf16x4(float4 v) {
  union { bf16 h[4]; uint2 u; } t;
  t.h[0] = (bf16)v.x; t.h[1] = (bf16)v.y; t.h[2] = (bf16)v.z; t.h[3] = (bf16)v.w;
  return t.u;
}

// Problem constants
constexpr int BATCH = 4;
constexpr int P     = 2048;
constexpr int D     = 1024;
constexpr int H     = 16;
constexpr int DK    = 64;
constexpr int NQKV  = H * DK;      // 1024

// ---------------------------------------------------------------------------
// Kernel 1: fused QKV projection.  Y = X(8192x1024) @ W(1024x1024) (+bias)
// Block tile 64x256, wave tile 32x64 (8 wmma/step), double-buffered LDS:
// next tile's global loads are in flight while current tile's wmma run.
//   Q -> [B,H,P,DK] (x 1/sqrt(DK)),  K -> [B,H,P,DK],  V -> [B,H,DK,P]
// ---------------------------------------------------------------------------
__global__ void __launch_bounds__(256)
qkv_kernel(const float* __restrict__ x,
           const float* __restrict__ Wq, const float* __restrict__ bq,
           const float* __restrict__ Wk, const float* __restrict__ bk,
           const float* __restrict__ Wv, const float* __restrict__ bv,
           bf16* __restrict__ Qb, bf16* __restrict__ Kb, bf16* __restrict__ Vt)
{
  __shared__ bf16 As[2][64][40];     // ping-pong 64x32 A tiles
  __shared__ bf16 Bs[2][256][48];    // ping-pong transposed [N][K] B tiles

  const int z  = blockIdx.z;
  const float* W    = (z == 0) ? Wq : ((z == 1) ? Wk : Wv);
  const float* bias = (z == 0) ? bq : ((z == 1) ? bk : bv);

  const int m0   = blockIdx.y * 64;
  const int n0   = blockIdx.x * 256;
  const int tid  = threadIdx.x;
  const int lane = tid & 31;
  const int wav  = tid >> 5;
  const int wm   = wav & 1;
  const int wn   = wav >> 1;
  const int half = lane >> 4;
  const int l16  = lane & 15;

  // staging geometry
  const int xrow = tid >> 3;          // 0..31
  const int xcol = (tid & 7) * 4;
  const int wkr  = tid >> 6;          // 0..3
  const int wcol = (tid & 63) * 4;

  float4 xr[2], wr[8];

  auto load_regs = [&](int k0) {
    #pragma unroll
    for (int r = 0; r < 2; ++r)
      xr[r] = *(const float4*)(x + (size_t)(m0 + xrow + 32 * r) * D + k0 + xcol);
    #pragma unroll
    for (int r = 0; r < 8; ++r)
      wr[r] = *(const float4*)(W + (size_t)(k0 + wkr + 4 * r) * NQKV + n0 + wcol);
    const int kp = (k0 + 32 < D) ? (k0 + 32) : k0;   // clamped, branch-free
    __builtin_prefetch(W + (size_t)(kp + wkr) * NQKV + n0 + wcol, 0, 1);
  };
  auto store_tile = [&](int bi) {
    #pragma unroll
    for (int r = 0; r < 2; ++r)
      *(uint2*)(&As[bi][xrow + 32 * r][xcol]) = pack_bf16x4(xr[r]);
    #pragma unroll
    for (int r = 0; r < 8; ++r) {
      Bs[bi][wcol + 0][wkr + 4 * r] = (bf16)wr[r].x;
      Bs[bi][wcol + 1][wkr + 4 * r] = (bf16)wr[r].y;
      Bs[bi][wcol + 2][wkr + 4 * r] = (bf16)wr[r].z;
      Bs[bi][wcol + 3][wkr + 4 * r] = (bf16)wr[r].w;
    }
  };

  v8f c[2][4] = {};

  load_regs(0);
  store_tile(0);
  constexpr int NK = D / 32;
  for (int t = 0; t < NK; ++t) {
    __syncthreads();
    const int cur = t & 1;
    if (t + 1 < NK) load_regs((t + 1) * 32);   // global loads overlap wmma below

    union { v16bf v; unsigned u[8]; } a[2];
    #pragma unroll
    for (int mi = 0; mi < 2; ++mi) {
      const bf16* Ar   = &As[cur][wm * 32 + mi * 16 + l16][0];
      const int   base = half * 16;
      #pragma unroll
      for (int e = 0; e < 16; e += 2) {
        const int k = base + e;
        a[mi].u[e >> 1] = *(const unsigned*)(Ar + k);
      }
    }
    #pragma unroll
    for (int j = 0; j < 4; ++j) {
      const v16bf bfrag = *(const v16bf*)(&Bs[cur][wn * 64 + j * 16 + l16][half * 16]);
      c[0][j] = WMMA_BF16(a[0].v, bfrag, c[0][j]);
      c[1][j] = WMMA_BF16(a[1].v, bfrag, c[1][j]);
    }
    if (t + 1 < NK) store_tile(cur ^ 1);       // waits loads after the wmma burst
  }

  // --- epilogue ---
  #pragma unroll
  for (int j = 0; j < 4; ++j) {
    const int   col = n0 + wn * 64 + j * 16 + l16;
    const float bb  = bias[col];
    const int   hh  = col >> 6;
    const int   dk  = col & 63;
    #pragma unroll
    for (int mi = 0; mi < 2; ++mi) {
      #pragma unroll
      for (int g = 0; g < 8; ++g) {
        const int mrow = m0 + wm * 32 + mi * 16 + g + 8 * half;
        float val = c[mi][j][g] + bb;
        if (z == 0) val *= 0.125f;                   // fold 1/sqrt(64) into Q
        const int bidx = mrow / P;
        const int p    = mrow % P;
        const bf16 bvv = (bf16)val;
        if (z == 2) {
          Vt[(((size_t)bidx * H + hh) * DK + dk) * P + p] = bvv;
        } else {
          bf16* dst = (z == 0) ? Qb : Kb;
          dst[(((size_t)bidx * H + hh) * P + p) * DK + dk] = bvv;
        }
      }
    }
  }
}

// ---------------------------------------------------------------------------
// Kernel 2: block-cooperative flash attention, double-buffered async staging.
// One block (8 waves) owns 128 queries of one (b,h); K/V 64x64 tiles ping-pong
// through LDS via async global->LDS copies while waves run 16 wmma per step.
// ---------------------------------------------------------------------------
__global__ void __launch_bounds__(256)
attn_kernel(const bf16* __restrict__ Qb, const bf16* __restrict__ Kb,
            const bf16* __restrict__ Vt, bf16* __restrict__ Rep)
{
  __shared__ bf16 Kl[2][64][72];
  __shared__ bf16 Vl[2][64][72];
  __shared__ bf16 Plds[8][16][72];

  const int tid  = threadIdx.x;
  const int lane = tid & 31;
  const int wav  = tid >> 5;
  const int half = lane >> 4;
  const int l16  = lane & 15;

  // blockIdx.x = ((b*H + h)*16 + qblk)
  const int qblk = blockIdx.x & 15;
  const int h    = (blockIdx.x >> 4) & 15;
  const int b    = blockIdx.x >> 8;
  const int q0   = qblk * 128 + wav * 16;

  const bf16* Qh = Qb + ((size_t)b * H + h) * P * DK;
  const bf16* Kh = Kb + ((size_t)b * H + h) * P * DK;
  const bf16* Vh = Vt + ((size_t)b * H + h) * DK * P;

  const int srow = tid >> 2;          // 0..63  (two 16B chunks per array each)
  const int scc  = (tid & 3) * 16;    // covers 64 cols in 4x16 chunks? no:
  // 64 cols = 8 chunks of 8 elems; 256 threads stage 512 chunks -> 2 each
  auto stage = [&](int kt, int bi) {
    #pragma unroll
    for (int i = 0; i < 2; ++i) {
      const int idx = tid + 256 * i;
      const int r   = idx >> 3;        // 0..63
      const int cc  = (idx & 7) * 8;   // 0..56
      copy16_to_lds(&Kl[bi][r][cc], Kh + (size_t)(kt + r) * DK + cc);
      copy16_to_lds(&Vl[bi][r][cc], Vh + (size_t)r * P + kt + cc);
    }
  };

  // Q A-fragments (dk 0..31 / 32..63), loaded once
  union { v16bf v; unsigned u[8]; } aq0, aq1;
  {
    const bf16* Qr   = Qh + (size_t)(q0 + l16) * DK;
    const int   base = half * 16;
    #pragma unroll
    for (int e = 0; e < 16; e += 2) {
      const int k = base + e;
      aq0.u[e >> 1] = *(const unsigned*)(Qr + k);
      aq1.u[e >> 1] = *(const unsigned*)(Qr + 32 + k);
    }
  }

  v8f   o[4] = {};
  float mrun[8], lrun[8];
  #pragma unroll
  for (int g = 0; g < 8; ++g) { mrun[g] = -1e30f; lrun[g] = 0.0f; }

  stage(0, 0);
  constexpr int NT = P / 64;
  for (int t = 0; t < NT; ++t) {
    ASYNC_WAIT();
    __syncthreads();
    const int cur = t & 1;
    if (t + 1 < NT) stage((t + 1) * 64, cur ^ 1);   // async, overlaps compute

    // ---- scores: 4 key sub-tiles x 2 dk-steps = 8 wmma ----
    v8f s[4] = {};
    #pragma unroll
    for (int nt = 0; nt < 4; ++nt) {
      const v16bf bk0 = *(const v16bf*)(&Kl[cur][nt * 16 + l16][half * 16]);
      const v16bf bk1 = *(const v16bf*)(&Kl[cur][nt * 16 + l16][32 + half * 16]);
      s[nt] = WMMA_BF16(aq0.v, bk0, s[nt]);
      s[nt] = WMMA_BF16(aq1.v, bk1, s[nt]);
    }

    // ---- online softmax over the 64-key window ----
    #pragma unroll
    for (int g = 0; g < 8; ++g) {
      float v = fmaxf(fmaxf(s[0][g], s[1][g]), fmaxf(s[2][g], s[3][g]));
      v = fmaxf(v, __shfl_xor(v, 1, 16));
      v = fmaxf(v, __shfl_xor(v, 2, 16));
      v = fmaxf(v, __shfl_xor(v, 4, 16));
      v = fmaxf(v, __shfl_xor(v, 8, 16));
      const float mnew = fmaxf(mrun[g], v);
      const float corr = __expf(mrun[g] - mnew);
      float p[4], rs = 0.0f;
      #pragma unroll
      for (int tt = 0; tt < 4; ++tt) { p[tt] = __expf(s[tt][g] - mnew); rs += p[tt]; }
      rs += __shfl_xor(rs, 1, 16);
      rs += __shfl_xor(rs, 2, 16);
      rs += __shfl_xor(rs, 4, 16);
      rs += __shfl_xor(rs, 8, 16);
      lrun[g] = lrun[g] * corr + rs;
      mrun[g] = mnew;
      #pragma unroll
      for (int j = 0; j < 4; ++j) o[j][g] *= corr;
      const int row = g + 8 * half;
      #pragma unroll
      for (int tt = 0; tt < 4; ++tt) Plds[wav][row][tt * 16 + l16] = (bf16)p[tt];
    }

    // ---- reload P-tile in A-layout (wave-private LDS) ----
    union { v16bf v; unsigned u[8]; } ap0, ap1;
    {
      const bf16* Pr   = &Plds[wav][l16][0];
      const int   base = half * 16;
      #pragma unroll
      for (int e = 0; e < 16; e += 2) {
        const int k = base + e;
        ap0.u[e >> 1] = *(const unsigned*)(Pr + k);
        ap1.u[e >> 1] = *(const unsigned*)(Pr + 32 + k);
      }
    }

    // ---- O += P * V : 4 dk tiles x 2 key-steps = 8 wmma ----
    #pragma unroll
    for (int j = 0; j < 4; ++j) {
      const v16bf bv0 = *(const v16bf*)(&Vl[cur][j * 16 + l16][half * 16]);
      const v16bf bv1 = *(const v16bf*)(&Vl[cur][j * 16 + l16][32 + half * 16]);
      o[j] = WMMA_BF16(ap0.v, bv0, o[j]);
      o[j] = WMMA_BF16(ap1.v, bv1, o[j]);
    }
  }

  // ---- normalize and store rep (bf16, [B*P, H*DK]) ----
  #pragma unroll
  for (int j = 0; j < 4; ++j) {
    #pragma unroll
    for (int g = 0; g < 8; ++g) {
      const int   row = q0 + g + 8 * half;
      const float val = o[j][g] / lrun[g];
      Rep[((size_t)b * P + row) * NQKV + h * 64 + j * 16 + l16] = (bf16)val;
    }
  }
  (void)srow; (void)scc;
}

// ---------------------------------------------------------------------------
// Kernel 3: output projection.  out = rep(bf16) @ Wo + bo  (fp32 out)
// Double-buffered: bf16 A tile via async global->LDS, Wo via register pipeline.
// ---------------------------------------------------------------------------
__global__ void __launch_bounds__(256)
out_proj_kernel(const bf16* __restrict__ Rep, const float* __restrict__ Wo,
                const float* __restrict__ bo, float* __restrict__ out)
{
  __shared__ bf16 As[2][64][40];
  __shared__ bf16 Bs[2][256][48];

  const int m0   = blockIdx.y * 64;
  const int n0   = blockIdx.x * 256;
  const int tid  = threadIdx.x;
  const int lane = tid & 31;
  const int wav  = tid >> 5;
  const int wm   = wav & 1;
  const int wn   = wav >> 1;
  const int half = lane >> 4;
  const int l16  = lane & 15;

  const int arow = tid >> 2;          // 0..63
  const int acc_ = (tid & 3) * 8;     // 0,8,16,24
  const int wkr  = tid >> 6;
  const int wcol = (tid & 63) * 4;

  float4 wr[8];

  auto stage_a = [&](int k0, int bi) {
    copy16_to_lds(&As[bi][arow][acc_], Rep + (size_t)(m0 + arow) * NQKV + k0 + acc_);
  };
  auto load_w = [&](int k0) {
    #pragma unroll
    for (int r = 0; r < 8; ++r)
      wr[r] = *(const float4*)(Wo + (size_t)(k0 + wkr + 4 * r) * D + n0 + wcol);
    const int kp = (k0 + 32 < NQKV) ? (k0 + 32) : k0;
    __builtin_prefetch(Wo + (size_t)(kp + wkr) * D + n0 + wcol, 0, 1);
  };
  auto store_w = [&](int bi) {
    #pragma unroll
    for (int r = 0; r < 8; ++r) {
      Bs[bi][wcol + 0][wkr + 4 * r] = (bf16)wr[r].x;
      Bs[bi][wcol + 1][wkr + 4 * r] = (bf16)wr[r].y;
      Bs[bi][wcol + 2][wkr + 4 * r] = (bf16)wr[r].z;
      Bs[bi][wcol + 3][wkr + 4 * r] = (bf16)wr[r].w;
    }
  };

  v8f c[2][4] = {};

  stage_a(0, 0);
  load_w(0);
  store_w(0);
  constexpr int NK = NQKV / 32;
  for (int t = 0; t < NK; ++t) {
    ASYNC_WAIT();
    __syncthreads();
    const int cur = t & 1;
    if (t + 1 < NK) { stage_a((t + 1) * 32, cur ^ 1); load_w((t + 1) * 32); }

    union { v16bf v; unsigned u[8]; } a[2];
    #pragma unroll
    for (int mi = 0; mi < 2; ++mi) {
      const bf16* Ar   = &As[cur][wm * 32 + mi * 16 + l16][0];
      const int   base = half * 16;
      #pragma unroll
      for (int e = 0; e < 16; e += 2) {
        const int k = base + e;
        a[mi].u[e >> 1] = *(const unsigned*)(Ar + k);
      }
    }
    #pragma unroll
    for (int j = 0; j < 4; ++j) {
      const v16bf bfrag = *(const v16bf*)(&Bs[cur][wn * 64 + j * 16 + l16][half * 16]);
      c[0][j] = WMMA_BF16(a[0].v, bfrag, c[0][j]);
      c[1][j] = WMMA_BF16(a[1].v, bfrag, c[1][j]);
    }
    if (t + 1 < NK) store_w(cur ^ 1);
  }

  #pragma unroll
  for (int j = 0; j < 4; ++j) {
    const int   col = n0 + wn * 64 + j * 16 + l16;
    const float bb  = bo[col];
    #pragma unroll
    for (int mi = 0; mi < 2; ++mi) {
      #pragma unroll
      for (int g = 0; g < 8; ++g) {
        const int mrow = m0 + wm * 32 + mi * 16 + g + 8 * half;
        out[(size_t)mrow * D + col] = c[mi][j][g] + bb;
      }
    }
  }
}

// ---------------------------------------------------------------------------
extern "C" void kernel_launch(void* const* d_in, const int* in_sizes, int n_in,
                              void* d_out, int out_size, void* d_ws, size_t ws_size,
                              hipStream_t stream) {
  (void)in_sizes; (void)n_in; (void)out_size; (void)ws_size;

  const float* x  = (const float*)d_in[0];
  const float* Wq = (const float*)d_in[1];
  const float* bq = (const float*)d_in[2];
  const float* Wk = (const float*)d_in[3];
  const float* bk = (const float*)d_in[4];
  const float* Wv = (const float*)d_in[5];
  const float* bv = (const float*)d_in[6];
  const float* Wo = (const float*)d_in[7];
  const float* bo = (const float*)d_in[8];
  float* out = (float*)d_out;

  const size_t qkv_elems = (size_t)BATCH * H * P * DK;   // 8,388,608 bf16 each
  bf16* Qb  = (bf16*)d_ws;
  bf16* Kb  = Qb + qkv_elems;
  bf16* Vt  = Kb + qkv_elems;
  bf16* Rep = Vt + qkv_elems;                            // total 64 MB of ws

  qkv_kernel<<<dim3(4, 128, 3), 256, 0, stream>>>(x, Wq, bq, Wk, bk, Wv, bv, Qb, Kb, Vt);
  attn_kernel<<<dim3(1024), 256, 0, stream>>>(Qb, Kb, Vt, Rep);
  out_proj_kernel<<<dim3(4, 128), 256, 0, stream>>>(Rep, Wo, bo, out);
}
